// MultiHeadAttention_4440996184818
// MI455X (gfx1250) — matrix operand order repaired and checked by reading, not compiled, not run on hardware
//
#include <hip/hip_runtime.h>
#include <hip/hip_bf16.h>

typedef __bf16 bf16;
typedef __attribute__((ext_vector_type(16))) __bf16 bf16x16;
typedef __attribute__((ext_vector_type(8)))  float  floatx8;
typedef __attribute__((ext_vector_type(4)))  int    i32x4;

#ifndef __has_builtin
#define __has_builtin(x) 0
#endif
#if __has_builtin(__builtin_amdgcn_global_load_async_to_lds_b128)
#define HAVE_ASYNC_LDS 1
#else
#define HAVE_ASYNC_LDS 0
#endif

typedef __attribute__((address_space(1))) i32x4* gi4p;   // global v4i32*
typedef __attribute__((address_space(3))) i32x4* li4p;   // LDS    v4i32*

// copy 16 contiguous bytes global -> LDS (async when available)
#if HAVE_ASYNC_LDS
#define ASYNC_CP16(g, l)                                                      \
  __builtin_amdgcn_global_load_async_to_lds_b128((gi4p)(g), (li4p)(l), 0, 0)
#else
#define ASYNC_CP16(g, l) (*(int4*)(l) = *(const int4*)(g))
#endif
#define WAIT_ASYNC() asm volatile("s_wait_asynccnt 0x0" ::: "memory")

union TileU { bf16x16 v; unsigned u[8]; };

// A-operand (16x32, 16-bit): lane L holds row m=L%16;
//   k = 16*(v/4) + 8*(L/16) + 2*(v%4) + h      (ISA 7.12.2 16-bit A table)
__device__ __forceinline__ bf16x16 load_a_tile(const bf16* p, int ld, int lane) {
  TileU t;
  const int row = lane & 15;
  const int kg  = (lane >> 4) << 3;              // 0 or 8
  const bf16* rp = p + (size_t)row * ld;
#pragma unroll
  for (int v = 0; v < 8; ++v) {
    const int k = ((v >> 2) << 4) + kg + ((v & 3) << 1);
    t.u[v] = *(const unsigned*)(rp + k);         // 2x b128 per lane
  }
  return t.v;
}

// B-operand (32x16, 16-bit) from Bt stored N-major (row=N, col=K):
//   lane L holds column n=L%16; k = 16*(v/4) + 8*(L/16) + 2*(v%4) + h
__device__ __forceinline__ bf16x16 load_b_tile(const bf16* p, int ld, int lane) {
  TileU t;
  const int row = lane & 15;                     // N index
  const int kg  = (lane >> 4) << 3;              // 0 or 8
  const bf16* rp = p + (size_t)row * ld + kg;
#pragma unroll
  for (int v = 0; v < 8; ++v) t.u[v] = *(const unsigned*)(rp + ((v >> 2) << 4) + ((v & 3) << 1));
  return t.v;
}

__device__ __forceinline__ floatx8 wmma_bf16(bf16x16 a, bf16x16 b, floatx8 c) {
  return __builtin_amdgcn_wmma_f32_16x16x32_bf16(false, a, false, b, (short)0, c,
                                                 false, false);
}

// ---------------------------------------------------------------------------
// bf16 GEMM: C[m,n] = sum_k A[m,k]*Bt[n,k].  Block = 8 waves -> 128x128 tile.
// B panel (128 x 32) staged in double-buffered LDS via async loads.
// grid = (M/128, N/128, batch)
// ---------------------------------------------------------------------------
#define BLD 56  // LDS row stride (elements): 112B -> bank-clean, 16B aligned

template <bool OUT_BF16>
__global__ __launch_bounds__(256)
void gemm_bt(const bf16* __restrict__ A, long sA, int lda,
             const bf16* __restrict__ Bt, long sB, int ldb,
             void* __restrict__ Cv, long sC, int ldc, int K)
{
  __shared__ __align__(16) bf16 ldsB[2][128 * BLD];
  const int tid  = threadIdx.x;
  const int lane = tid & 31;
  const int wave = tid >> 5;
  const int wm = wave & 3, wn = wave >> 2;
  const int m0  = blockIdx.x * 128 + wm * 32;
  const int n0b = blockIdx.y * 128;
  const int n0w = wn * 64;
  const int batch = blockIdx.z;
  const bf16* a  = A  + (size_t)batch * sA;
  const bf16* bt = Bt + (size_t)batch * sB;

  const int crow = tid >> 2;             // 0..63
  const int cchk = (tid & 3) * 8;        // element offset: 0/8/16/24
  auto copyB = [&](int buf, int k0) {
#pragma unroll
    for (int it = 0; it < 2; ++it) {
      const int r = crow + it * 64;
      const bf16* g = bt + (size_t)(n0b + r) * ldb + k0 + cchk;
      bf16* l = &ldsB[buf][r * BLD + cchk];
      ASYNC_CP16(g, l);
    }
  };

  floatx8 acc[2][4];
#pragma unroll
  for (int i = 0; i < 2; ++i)
#pragma unroll
    for (int t = 0; t < 4; ++t) acc[i][t] = (floatx8)0.0f;

  copyB(0, 0);
  const int nsteps = K / 32;
  for (int st = 0; st < nsteps; ++st) {
    const int k0 = st * 32;
    WAIT_ASYNC();
    __syncthreads();                         // panel st ready for all waves
    if (st + 1 < nsteps) copyB((st + 1) & 1, k0 + 32);
    if (st + 1 < nsteps)
      __builtin_prefetch(a + (size_t)m0 * lda + k0 + 32, 0, 1);

    bf16x16 at[2];
    at[0] = load_a_tile(a + (size_t)(m0     ) * lda + k0, lda, lane);
    at[1] = load_a_tile(a + (size_t)(m0 + 16) * lda + k0, lda, lane);
    bf16x16 btl[4];
#pragma unroll
    for (int t = 0; t < 4; ++t)
      btl[t] = load_b_tile(&ldsB[st & 1][(n0w + 16 * t) * BLD], BLD, lane);
#pragma unroll
    for (int i = 0; i < 2; ++i)
#pragma unroll
      for (int t = 0; t < 4; ++t)
        acc[i][t] = wmma_bf16(at[i], btl[t], acc[i][t]);
  }

  const int cn = lane & 15;
  const int mg = (lane >> 4) << 3;
#pragma unroll
  for (int i = 0; i < 2; ++i)
#pragma unroll
    for (int t = 0; t < 4; ++t)
#pragma unroll
      for (int r = 0; r < 8; ++r) {
        const size_t m  = (size_t)(m0 + 16 * i + mg + r);
        const int    nn = n0b + n0w + 16 * t + cn;
        const float val = acc[i][t][r];
        if (OUT_BF16)
          ((bf16*)Cv + (size_t)batch * sC)[m * ldc + nn] = (bf16)val;
        else
          ((float*)Cv + (size_t)batch * sC)[m * ldc + nn] = val;
      }
}

// ---------------------------------------------------------------------------
// Flash attention: block = 8 waves = 128 query rows of one (head, batch).
// K (32x64) / V (64x32) j-block tiles async-staged into shared LDS
// (double-buffered) -> 8x less L2 traffic than per-wave fetching.
// grid = (seq/128, H, B), block = 256.
// ---------------------------------------------------------------------------
#define KLD 72  // 144B rows: bank-clean, 16B aligned
#define VLD 56  // 112B rows
#define PLD 48  //  96B rows (per-wave P staging)

__global__ __launch_bounds__(256)
void attn_flash(const bf16* __restrict__ QT, const bf16* __restrict__ KT,
                const bf16* __restrict__ V, bf16* __restrict__ YT,
                int seq, int C)
{
  __shared__ __align__(16) bf16 ldsK[2][32 * KLD];
  __shared__ __align__(16) bf16 ldsV[2][64 * VLD];
  __shared__ __align__(16) bf16 ldsP[8][16 * PLD];

  const int tid  = threadIdx.x;
  const int lane = tid & 31;
  const int wave = tid >> 5;
  const int h     = blockIdx.y;
  const int batch = blockIdx.z;
  const int i0    = blockIdx.x * 128 + wave * 16;
  const size_t bqk = (size_t)batch * seq * C;
  const bf16* qt = QT + bqk;
  const bf16* kt = KT + bqk;
  const bf16* vv = V + (size_t)batch * C * seq;   // (C x seq)
  bf16* yt = YT + bqk;
  const int hd = h * 64;

  const int krow_c = tid >> 3, kchk_c = (tid & 7) * 8;  // K: 32 rows x 128B
  const int vrow_c = tid >> 2, vchk_c = (tid & 3) * 8;  // V: 64 rows x 64B
  auto copyKV = [&](int buf, int j0) {
    {
      const bf16* g = kt + (size_t)(j0 + krow_c) * C + hd + kchk_c;
      bf16* l = &ldsK[buf][krow_c * KLD + kchk_c];
      ASYNC_CP16(g, l);
    }
    {
      const bf16* g = vv + (size_t)(hd + vrow_c) * seq + j0 + vchk_c;
      bf16* l = &ldsV[buf][vrow_c * VLD + vchk_c];
      ASYNC_CP16(g, l);
    }
  };

  bf16x16 qa[2];
  qa[0] = load_a_tile(qt + (size_t)i0 * C + hd,      C, lane);
  qa[1] = load_a_tile(qt + (size_t)i0 * C + hd + 32, C, lane);

  float mrow[8], lrow[8];
#pragma unroll
  for (int r = 0; r < 8; ++r) { mrow[r] = -1e30f; lrow[r] = 0.0f; }
  floatx8 o[4];
#pragma unroll
  for (int t = 0; t < 4; ++t) o[t] = (floatx8)0.0f;

  const int mg = (lane >> 4) << 3;
  const int cn = lane & 15;
  bf16* pl = &ldsP[wave][0];

  copyKV(0, 0);
  const int nsteps = seq / 32;
  for (int st = 0; st < nsteps; ++st) {
    const int buf = st & 1;
    WAIT_ASYNC();
    __syncthreads();                         // K/V tiles for step st ready
    if (st + 1 < nsteps) copyKV(buf ^ 1, (st + 1) * 32);

    // S = Q.K^T over d=64 (two K=32 chunks), two 16-col tiles
    floatx8 s[2];
#pragma unroll
    for (int jt = 0; jt < 2; ++jt) {
      bf16x16 kb0 = load_b_tile(&ldsK[buf][(16 * jt) * KLD],      KLD, lane);
      bf16x16 kb1 = load_b_tile(&ldsK[buf][(16 * jt) * KLD + 32], KLD, lane);
      floatx8 z = (floatx8)0.0f;
      z     = wmma_bf16(qa[0], kb0, z);
      s[jt] = wmma_bf16(qa[1], kb1, z);
    }

    // online softmax; mask-outer so all 8 row butterflies issue per level
    float mx[8];
#pragma unroll
    for (int r = 0; r < 8; ++r) mx[r] = fmaxf(s[0][r], s[1][r]);
#pragma unroll
    for (int msk = 8; msk >= 1; msk >>= 1)
#pragma unroll
      for (int r = 0; r < 8; ++r)
        mx[r] = fmaxf(mx[r], __shfl_xor(mx[r], msk, 32));
    float alpha[8];
#pragma unroll
    for (int r = 0; r < 8; ++r) {
      const float mnew = fmaxf(mrow[r], mx[r]);
      alpha[r] = __expf(mrow[r] - mnew);
      mrow[r]  = mnew;
    }
    floatx8 p[2];
#pragma unroll
    for (int jt = 0; jt < 2; ++jt)
#pragma unroll
      for (int r = 0; r < 8; ++r)
        p[jt][r] = __expf(s[jt][r] - mrow[r]);
    float sm[8];
#pragma unroll
    for (int r = 0; r < 8; ++r) sm[r] = p[0][r] + p[1][r];
#pragma unroll
    for (int msk = 8; msk >= 1; msk >>= 1)
#pragma unroll
      for (int r = 0; r < 8; ++r)
        sm[r] += __shfl_xor(sm[r], msk, 32);
#pragma unroll
    for (int r = 0; r < 8; ++r) lrow[r] = lrow[r] * alpha[r] + sm[r];
#pragma unroll
    for (int t = 0; t < 4; ++t)
#pragma unroll
      for (int r = 0; r < 8; ++r) o[t][r] *= alpha[r];

    // C/D-layout P -> per-wave LDS -> reload in A-operand layout
#pragma unroll
    for (int jt = 0; jt < 2; ++jt)
#pragma unroll
      for (int r = 0; r < 8; ++r)
        pl[(mg + r) * PLD + jt * 16 + cn] = (bf16)p[jt][r];
    asm volatile("" ::: "memory");           // DS is in-order within a wave
    bf16x16 pa = load_a_tile(pl, PLD, lane);
    asm volatile("" ::: "memory");

    // O += P.V_blk  (K=32, four 16-wide d tiles)
#pragma unroll
    for (int t = 0; t < 4; ++t) {
      bf16x16 vb = load_b_tile(&ldsV[buf][(16 * t) * VLD], VLD, lane);
      o[t] = wmma_bf16(pa, vb, o[t]);
    }
  }

#pragma unroll
  for (int t = 0; t < 4; ++t)
#pragma unroll
    for (int r = 0; r < 8; ++r) {
      const size_t m = (size_t)(i0 + mg + r);
      yt[m * C + hd + 16 * t + cn] = (bf16)(o[t][r] / lrow[r]);
    }
}

// ---------------------------------------------------------------------------
__global__ void cvt_bf16(const float* __restrict__ s, bf16* __restrict__ d, int count) {
  int i = blockIdx.x * blockDim.x + threadIdx.x;
  if (i < count) d[i] = (bf16)s[i];
}

__global__ __launch_bounds__(256)
void xpose_bf16(const float* __restrict__ x, bf16* __restrict__ xT, int C, int seq) {
  __shared__ float tile[32][33];
  const int tx = threadIdx.x, ty = threadIdx.y;  // block (32, 8)
  const int n0 = blockIdx.x * 32;
  const int c0 = blockIdx.y * 32;
  const int batch = blockIdx.z;
  const float* xb = x + (size_t)batch * C * seq;
  bf16* xtb = xT + (size_t)batch * seq * C;
#pragma unroll
  for (int r = 0; r < 4; ++r)
    tile[ty + 8 * r][tx] = xb[(size_t)(c0 + ty + 8 * r) * seq + n0 + tx];
  __syncthreads();
#pragma unroll
  for (int r = 0; r < 4; ++r)
    xtb[(size_t)(n0 + ty + 8 * r) * C + c0 + tx] = (bf16)tile[tx][ty + 8 * r];
}

// ---------------------------------------------------------------------------
extern "C" void kernel_launch(void* const* d_in, const int* in_sizes, int n_in,
                              void* d_out, int out_size, void* d_ws, size_t ws_size,
                              hipStream_t stream) {
  (void)in_sizes; (void)n_in; (void)out_size; (void)ws_size;
  const int B = 4, C = 512, SEQ = 1024, H = 8;
  const float* x  = (const float*)d_in[0];
  const float* Wq = (const float*)d_in[1];
  const float* Wk = (const float*)d_in[2];
  const float* Wv = (const float*)d_in[3];
  const float* Wo = (const float*)d_in[4];
  float* out = (float*)d_out;

  char* ws = (char*)d_ws;
  size_t off = 0;
  auto alc = [&](size_t bytes) -> void* {
    void* p = ws + off;
    off += (bytes + 255) & ~(size_t)255;
    return p;
  };
  bf16* wq = (bf16*)alc((size_t)C * C * 2);
  bf16* wk = (bf16*)alc((size_t)C * C * 2);
  bf16* wv = (bf16*)alc((size_t)C * C * 2);
  bf16* wo = (bf16*)alc((size_t)C * C * 2);
  bf16* xT = (bf16*)alc((size_t)B * SEQ * C * 2);
  bf16* QT = (bf16*)alc((size_t)B * SEQ * C * 2);
  bf16* KT = (bf16*)alc((size_t)B * SEQ * C * 2);
  bf16* Vb = (bf16*)alc((size_t)B * C * SEQ * 2);
  bf16* YT = (bf16*)alc((size_t)B * SEQ * C * 2);

  const int wcnt = C * C;
  cvt_bf16<<<(wcnt + 255) / 256, 256, 0, stream>>>(Wq, wq, wcnt);
  cvt_bf16<<<(wcnt + 255) / 256, 256, 0, stream>>>(Wk, wk, wcnt);
  cvt_bf16<<<(wcnt + 255) / 256, 256, 0, stream>>>(Wv, wv, wcnt);
  cvt_bf16<<<(wcnt + 255) / 256, 256, 0, stream>>>(Wo, wo, wcnt);

  xpose_bf16<<<dim3(SEQ / 32, C / 32, B), dim3(32, 8), 0, stream>>>(x, xT, C, SEQ);

  const long sXT = (long)SEQ * C;
  const long sV  = (long)C * SEQ;

  // Q^T = x^T . Wq^T  (M=seq, N=C, K=C)
  gemm_bt<true><<<dim3(SEQ / 128, C / 128, B), 256, 0, stream>>>(
      xT, sXT, C, wq, 0, C, QT, sXT, C, C);
  // K^T = x^T . Wk^T
  gemm_bt<true><<<dim3(SEQ / 128, C / 128, B), 256, 0, stream>>>(
      xT, sXT, C, wk, 0, C, KT, sXT, C, C);
  // V = Wv . x  (M=C, N=seq, K=C), stored (C x seq)
  gemm_bt<true><<<dim3(C / 128, SEQ / 128, B), 256, 0, stream>>>(
      wv, 0, C, xT, sXT, C, Vb, sV, SEQ, C);

  // flash attention -> Y^T (seq x C)
  attn_flash<<<dim3(SEQ / 128, H, B), 256, 0, stream>>>(QT, KT, Vb, YT, SEQ, C);

  // out = Wo . y  (M=C, N=seq, K=C) fp32 -> d_out (B, C, seq)
  gemm_bt<false><<<dim3(C / 128, SEQ / 128, B), 256, 0, stream>>>(
      wo, 0, C, YT, sXT, C, out, sV, SEQ, C);
}
